// MultiHeadAttention_11630771438155
// MI455X (gfx1250) — matrix operand order repaired and checked by reading, not compiled, not run on hardware
//
#include <hip/hip_runtime.h>

// ---------------------------------------------------------------------------
// MultiHeadAttention block (proj -> attention+softmax -> out proj -> LN)
// MI455X / gfx1250, wave32, V_WMMA_F32_16X16X32_F16 + async global->LDS.
// ---------------------------------------------------------------------------

typedef _Float16 v16h __attribute__((ext_vector_type(16)));
typedef _Float16 v8h  __attribute__((ext_vector_type(8)));
typedef float    v8f  __attribute__((ext_vector_type(8)));

#define B_   2
#define S_   2048
#define DM_  1024
#define NH_  16
#define DK_  64
#define MTOK (B_*S_)          // 4096 tokens
#define NEG_ (-1.0e9f)

// gfx1250 async global->LDS (ASYNCcnt path). GVS mode: SGPR64 base + VGPR32 offset.
#define ASYNC_LDS_B128(ldsoff, goff, base64)                                   \
  asm volatile("global_load_async_to_lds_b128 %0, %1, %2 offset:0"             \
               :: "v"(ldsoff), "v"(goff),                                      \
                  "s"((unsigned long long)(size_t)(base64)) : "memory")
#define WAIT_ASYNC() asm volatile("s_wait_asynccnt 0x0" ::: "memory")
#define WAIT_DS()    asm volatile("s_wait_dscnt 0x0" ::: "memory")

// --- helpers ---------------------------------------------------------------

__device__ __forceinline__ v16h cat8(v8h lo, v8h hi) {
  return __builtin_shufflevector(lo, hi, 0,1,2,3,4,5,6,7,8,9,10,11,12,13,14,15);
}

__device__ __forceinline__ v8f wmma_f16(v16h a, v16h b, v8f c) {
  // D = A(16x32 f16) * B(32x16 f16) + C(16x16 f32)
  return __builtin_amdgcn_wmma_f32_16x16x32_f16(false, a, false, b, (short)0, c,
                                                false, false);
}

// A fragment 16x32 f16 from row-major source.
// Layout: lane<16 holds row=lr, K in {0..15}; lane>=16 K in {16..31}
__device__ __forceinline__ v16h load_a(const _Float16* base, int lda, int lr,
                                       int k0, int hh) {
  const _Float16* p = base + (size_t)lr * lda + k0 + hh * 16;
  v8h lo = *(const v8h*)p;
  v8h hi = *(const v8h*)(p + 8);
  return cat8(lo, hi);
}

// B fragment 32x16 f16. Source row n holds K contiguous (W^T[N][K] layout).
__device__ __forceinline__ v16h load_b(const _Float16* base, int ldb, int lr,
                                       int k0, int hh) {
  return *(const v16h*)(base + (size_t)lr * ldb + k0 + hh * 16);
}

// --- conversion / packing kernels ------------------------------------------

__global__ __launch_bounds__(256) void cvt_f32_f16(const float* __restrict__ src,
                                                   _Float16* __restrict__ dst, int n) {
  int i = blockIdx.x * 256 + threadIdx.x;
  if (i < n) dst[i] = (_Float16)src[i];
}

// wt[n][k] = w[k][n]
__global__ __launch_bounds__(256) void cvt_wT(const float* __restrict__ w,
                                              _Float16* __restrict__ wt) {
  int i = blockIdx.x * 256 + threadIdx.x;     // i = k*1024 + n
  int k = i >> 10, n = i & 1023;
  wt[(size_t)n * DM_ + k] = (_Float16)w[i];
}

// pack bool mask into bit matrix: pm[b][row][w] bit c = mask[b][row][32w+c]
__global__ __launch_bounds__(256) void pack_mask(const unsigned char* __restrict__ mask,
                                                 unsigned* __restrict__ pm) {
  int w = blockIdx.x * 256 + threadIdx.x;     // 0 .. B*S*64-1
  const uint4* p4 = (const uint4*)(mask + (size_t)w * 32);
  uint4 a = p4[0], bq = p4[1];
  unsigned ww[8] = {a.x, a.y, a.z, a.w, bq.x, bq.y, bq.z, bq.w};
  unsigned bits = 0;
  #pragma unroll
  for (int j = 0; j < 8; j++)
    #pragma unroll
    for (int c = 0; c < 4; c++)
      if ((ww[j] >> (8 * c)) & 0xffu) bits |= 1u << (j * 4 + c);
  pm[w] = bits;
}

// --- projection GEMM: Y[4096x1024] = X16 @ W^T ------------------------------
// MODE 0: Y layout [B,H,S,64] (Q,K).  MODE 1: Y layout [B,H,64,S] (V transposed).

template <int MODE>
__global__ __launch_bounds__(256) void proj_gemm(const _Float16* __restrict__ X,
                                                 const _Float16* __restrict__ Wt,
                                                 _Float16* __restrict__ Y) {
  int wid = threadIdx.x >> 5, lane = threadIdx.x & 31;
  int hh = lane >> 4, lr = lane & 15;
  int tile = blockIdx.x * 8 + wid;            // 1024 tiles: 64 (M) x 16 (N)
  int m0 = (tile & 63) * 64, n0 = (tile >> 6) * 64;

  v8f acc[4][4] = {};
  for (int k0 = 0; k0 < DM_; k0 += 32) {
    v16h a[4], b[4];
    #pragma unroll
    for (int mi = 0; mi < 4; mi++)
      a[mi] = load_a(X + (size_t)(m0 + 16 * mi) * DM_, DM_, lr, k0, hh);
    #pragma unroll
    for (int ni = 0; ni < 4; ni++)
      b[ni] = load_b(Wt + (size_t)(n0 + 16 * ni) * DM_, DM_, lr, k0, hh);
    #pragma unroll
    for (int mi = 0; mi < 4; mi++)
      #pragma unroll
      for (int ni = 0; ni < 4; ni++)
        acc[mi][ni] = wmma_f16(a[mi], b[ni], acc[mi][ni]);
  }

  #pragma unroll
  for (int mi = 0; mi < 4; mi++)
    #pragma unroll
    for (int ni = 0; ni < 4; ni++) {
      if (MODE == 1) {
        // lane's 8 C-elements are contiguous tokens at fixed d -> one B128 store
        int m = m0 + 16 * mi + 8 * hh;
        int n = n0 + 16 * ni + lr;
        int bb = m >> 11, s0 = m & (S_ - 1);
        int h = n >> 6, d = n & 63;
        v8h pk;
        #pragma unroll
        for (int i = 0; i < 8; i++) pk[i] = (_Float16)acc[mi][ni][i];
        *(v8h*)&Y[(((size_t)bb * NH_ + h) * DK_ + d) * S_ + s0] = pk;
      } else {
        #pragma unroll
        for (int i = 0; i < 8; i++) {
          int m = m0 + 16 * mi + 8 * hh + i;
          int n = n0 + 16 * ni + lr;
          int bb = m >> 11, s = m & (S_ - 1);
          int h = n >> 6, d = n & 63;
          Y[(((size_t)bb * NH_ + h) * S_ + s) * DK_ + d] = (_Float16)acc[mi][ni][i];
        }
      }
    }
}

// --- attention kernel -------------------------------------------------------
// One block = one (b,h, 128-query chunk); 8 waves x 16-query tiles.
// K/V 32-key tiles staged cooperatively into LDS with global_load_async_to_lds.
// pass1: flash-style online max/sum; pass2: recompute, write attn once,
// LDS-transpose probs into A-fragments, accumulate ctx = P @ V via WMMA.

__global__ __launch_bounds__(256) void attn_kernel(const _Float16* __restrict__ qh,
                                                   const _Float16* __restrict__ kh,
                                                   const _Float16* __restrict__ vt,
                                                   const unsigned* __restrict__ pm,
                                                   float* __restrict__ attn,
                                                   _Float16* __restrict__ ctx16) {
  __shared__ __attribute__((aligned(32))) _Float16 kbuf[32][72]; // 32 keys x 64 d (+pad)
  __shared__ __attribute__((aligned(32))) _Float16 vbuf[64][40]; // 64 d x 32 keys (+pad)
  __shared__ __attribute__((aligned(32))) _Float16 sc[8][16][40];// per-wave prob tiles

  int tid = threadIdx.x;
  int wid = tid >> 5, lane = tid & 31;
  int hh = lane >> 4, lr = lane & 15;
  int bh = blockIdx.x >> 4, chunk = blockIdx.x & 15;
  int b = bh >> 4, h = bh & 15;
  int q0 = chunk * 128 + wid * 16;

  const _Float16* Q  = qh + ((size_t)bh * S_ + q0) * DK_;
  const _Float16* Kp = kh + (size_t)bh * S_ * DK_;
  const _Float16* Vp = vt + (size_t)bh * DK_ * S_;

  // cooperative staging roles (256 threads x 16B = 4KB per tile)
  int krow = tid >> 3, kcol = (tid & 7) * 8;           // kbuf[32][64]
  unsigned klds = (unsigned)(size_t)&kbuf[krow][kcol];
  unsigned kgo  = (unsigned)(krow * (DK_ * 2) + kcol * 2);
  int vrow = tid >> 2, vcol = (tid & 3) * 8;           // vbuf[64][32]
  unsigned vlds = (unsigned)(size_t)&vbuf[vrow][vcol];
  unsigned vgo  = (unsigned)(vrow * (S_ * 2) + vcol * 2);

  v16h qa[2];
  #pragma unroll
  for (int j = 0; j < 2; j++) qa[j] = load_a(Q, DK_, lr, j * 32, hh);

  float m[8], s[8];
  #pragma unroll
  for (int i = 0; i < 8; i++) { m[i] = -3.0e38f; s[i] = 0.f; }

  // ---- pass 1: online softmax statistics ----
  for (int n0 = 0; n0 < S_; n0 += 32) {
    __syncthreads();                                    // kbuf safe to overwrite
    ASYNC_LDS_B128(klds, kgo, Kp + (size_t)n0 * DK_);
    WAIT_ASYNC();
    __syncthreads();                                    // tile visible to all waves
    v8f c0 = {}, c1 = {};
    #pragma unroll
    for (int j = 0; j < 2; j++) {
      v16h b0 = *(const v16h*)&kbuf[lr][j * 32 + hh * 16];
      v16h b1 = *(const v16h*)&kbuf[16 + lr][j * 32 + hh * 16];
      c0 = wmma_f16(qa[j], b0, c0);
      c1 = wmma_f16(qa[j], b1, c1);
    }
    #pragma unroll
    for (int i = 0; i < 8; i++) {
      int ri = q0 + 8 * hh + i;
      unsigned mw = pm[(((size_t)b * S_ + ri) << 6) + (n0 >> 5)];
      float x0 = ((mw >> lr) & 1u)        ? NEG_ : c0[i] * 0.125f;
      float x1 = ((mw >> (16 + lr)) & 1u) ? NEG_ : c1[i] * 0.125f;
      float mn = fmaxf(m[i], fmaxf(x0, x1));
      s[i] = s[i] * __expf(m[i] - mn) + __expf(x0 - mn) + __expf(x1 - mn);
      m[i] = mn;
    }
  }
  // combine across the 16 lanes sharing each row
  #pragma unroll
  for (int off = 1; off < 16; off <<= 1) {
    #pragma unroll
    for (int i = 0; i < 8; i++) {
      float om = __shfl_xor(m[i], off, 32);
      float os = __shfl_xor(s[i], off, 32);
      float mn = fmaxf(m[i], om);
      s[i] = s[i] * __expf(m[i] - mn) + os * __expf(om - mn);
      m[i] = mn;
    }
  }
  float rs[8];
  #pragma unroll
  for (int i = 0; i < 8; i++) rs[i] = 1.0f / s[i];

  // ---- pass 2: write attn once, accumulate ctx = P @ V ----
  v8f ctx[4] = {};
  for (int n0 = 0; n0 < S_; n0 += 32) {
    __syncthreads();
    ASYNC_LDS_B128(klds, kgo, Kp + (size_t)n0 * DK_);
    ASYNC_LDS_B128(vlds, vgo, Vp + n0);
    WAIT_ASYNC();
    __syncthreads();
    v8f c0 = {}, c1 = {};
    #pragma unroll
    for (int j = 0; j < 2; j++) {
      v16h b0 = *(const v16h*)&kbuf[lr][j * 32 + hh * 16];
      v16h b1 = *(const v16h*)&kbuf[16 + lr][j * 32 + hh * 16];
      c0 = wmma_f16(qa[j], b0, c0);
      c1 = wmma_f16(qa[j], b1, c1);
    }
    #pragma unroll
    for (int i = 0; i < 8; i++) {
      int r8 = 8 * hh + i;
      int ri = q0 + r8;
      unsigned mw = pm[(((size_t)b * S_ + ri) << 6) + (n0 >> 5)];
      float x0 = ((mw >> lr) & 1u)        ? NEG_ : c0[i] * 0.125f;
      float x1 = ((mw >> (16 + lr)) & 1u) ? NEG_ : c1[i] * 0.125f;
      float p0 = __expf(x0 - m[i]) * rs[i];
      float p1 = __expf(x1 - m[i]) * rs[i];
      float* arow = attn + ((size_t)bh * S_ + ri) * S_;
      arow[n0 + lr]      = p0;              // 16 consecutive floats per half-wave
      arow[n0 + 16 + lr] = p1;
      sc[wid][r8][lr]      = (_Float16)p0;  // C-layout -> LDS prob tile
      sc[wid][r8][16 + lr] = (_Float16)p1;
    }
    WAIT_DS();                              // per-wave cross-lane LDS visibility
    const _Float16* sp = &sc[wid][lr][hh * 16];
    v16h pa = cat8(*(const v8h*)sp, *(const v8h*)(sp + 8));
    #pragma unroll
    for (int ni = 0; ni < 4; ni++) {
      v16h vb = *(const v16h*)&vbuf[16 * ni + lr][hh * 16];
      ctx[ni] = wmma_f16(pa, vb, ctx[ni]);
    }
  }

  // store ctx into [token][h*64+d] f16 for the output projection GEMM
  #pragma unroll
  for (int ni = 0; ni < 4; ni++)
    #pragma unroll
    for (int i = 0; i < 8; i++) {
      int tok = b * S_ + q0 + 8 * hh + i;
      int col = h * DK_ + 16 * ni + lr;
      ctx16[(size_t)tok * DM_ + col] = (_Float16)ctx[ni][i];
    }
}

// --- output projection GEMM + residual -> xres (f32) ------------------------

__global__ __launch_bounds__(256) void fc_gemm(const _Float16* __restrict__ Ctx,
                                               const _Float16* __restrict__ Wt,
                                               const float* __restrict__ resid,
                                               float* __restrict__ Xres) {
  int wid = threadIdx.x >> 5, lane = threadIdx.x & 31;
  int hh = lane >> 4, lr = lane & 15;
  int tile = blockIdx.x * 8 + wid;
  int m0 = (tile & 63) * 64, n0 = (tile >> 6) * 64;

  v8f acc[4][4] = {};
  for (int k0 = 0; k0 < DM_; k0 += 32) {
    v16h a[4], b[4];
    #pragma unroll
    for (int mi = 0; mi < 4; mi++)
      a[mi] = load_a(Ctx + (size_t)(m0 + 16 * mi) * DM_, DM_, lr, k0, hh);
    #pragma unroll
    for (int ni = 0; ni < 4; ni++)
      b[ni] = load_b(Wt + (size_t)(n0 + 16 * ni) * DM_, DM_, lr, k0, hh);
    #pragma unroll
    for (int mi = 0; mi < 4; mi++)
      #pragma unroll
      for (int ni = 0; ni < 4; ni++)
        acc[mi][ni] = wmma_f16(a[mi], b[ni], acc[mi][ni]);
  }
  #pragma unroll
  for (int mi = 0; mi < 4; mi++)
    #pragma unroll
    for (int ni = 0; ni < 4; ni++)
      #pragma unroll
      for (int i = 0; i < 8; i++) {
        int m = m0 + 16 * mi + 8 * hh + i;
        int n = n0 + 16 * ni + lr;
        size_t idx = (size_t)m * DM_ + n;
        Xres[idx] = acc[mi][ni][i] + resid[idx];
      }
}

// --- layernorm --------------------------------------------------------------

__global__ __launch_bounds__(256) void ln_kernel(const float* __restrict__ X,
                                                 const float* __restrict__ gamma,
                                                 const float* __restrict__ beta,
                                                 float* __restrict__ out) {
  int row = blockIdx.x;
  const float* xr = X + (size_t)row * DM_;
  float v[4], s1 = 0.f, s2 = 0.f;
  #pragma unroll
  for (int j = 0; j < 4; j++) {
    v[j] = xr[threadIdx.x + j * 256];
    s1 += v[j];
    s2 += v[j] * v[j];
  }
  #pragma unroll
  for (int off = 16; off > 0; off >>= 1) {
    s1 += __shfl_xor(s1, off, 32);
    s2 += __shfl_xor(s2, off, 32);
  }
  __shared__ float red1[8], red2[8];
  int wid = threadIdx.x >> 5, lane = threadIdx.x & 31;
  if (lane == 0) { red1[wid] = s1; red2[wid] = s2; }
  __syncthreads();
  float t1 = 0.f, t2 = 0.f;
  #pragma unroll
  for (int w = 0; w < 8; w++) { t1 += red1[w]; t2 += red2[w]; }
  float mu = t1 * (1.0f / DM_);
  float var = t2 * (1.0f / DM_) - mu * mu;
  float r = rsqrtf(var + 1e-5f);
  float* orow = out + (size_t)row * DM_;
  #pragma unroll
  for (int j = 0; j < 4; j++) {
    int c = threadIdx.x + j * 256;
    orow[c] = (v[j] - mu) * r * gamma[c] + beta[c];
  }
}

// --- launch ------------------------------------------------------------------

extern "C" void kernel_launch(void* const* d_in, const int* in_sizes, int n_in,
                              void* d_out, int out_size, void* d_ws, size_t ws_size,
                              hipStream_t stream) {
  const float* q  = (const float*)d_in[0];
  const float* k  = (const float*)d_in[1];
  const float* v  = (const float*)d_in[2];
  const unsigned char* mask = (const unsigned char*)d_in[3];
  const float* w_q  = (const float*)d_in[4];
  const float* w_k  = (const float*)d_in[5];
  const float* w_v  = (const float*)d_in[6];
  const float* w_fc = (const float*)d_in[7];
  const float* gamma = (const float*)d_in[8];
  const float* beta  = (const float*)d_in[9];

  float* out  = (float*)d_out;                       // [B,S,1024]
  float* attn = out + (size_t)MTOK * DM_;            // [B,H,S,S]

  // workspace map (needs 57 MB); ctx16/xres alias dead regions.
  char* ws = (char*)d_ws;
  const size_t MB = 1u << 20;
  _Float16* q16   = (_Float16*)(ws + 0 * MB);
  _Float16* k16   = (_Float16*)(ws + 8 * MB);
  _Float16* v16   = (_Float16*)(ws + 16 * MB);
  _Float16* wqT   = (_Float16*)(ws + 24 * MB);
  _Float16* wkT   = (_Float16*)(ws + 26 * MB);
  _Float16* wvT   = (_Float16*)(ws + 28 * MB);
  _Float16* wfcT  = (_Float16*)(ws + 30 * MB);
  _Float16* qh    = (_Float16*)(ws + 32 * MB);       // [B,H,S,64]
  _Float16* kh    = (_Float16*)(ws + 40 * MB);       // [B,H,S,64]
  _Float16* vt    = (_Float16*)(ws + 48 * MB);       // [B,H,64,S]
  unsigned* pmask = (unsigned*)(ws + 56 * MB);       // packed mask bits (1MB)
  _Float16* ctx16 = (_Float16*)(ws + 0 * MB);        // aliases q16 (dead)
  float*    xres  = (float*)   (ws + 8 * MB);        // aliases k16/v16 (dead)

  const int nx = MTOK * DM_;                         // 4,194,304
  cvt_f32_f16<<<nx / 256, 256, 0, stream>>>(q, q16, nx);
  cvt_f32_f16<<<nx / 256, 256, 0, stream>>>(k, k16, nx);
  cvt_f32_f16<<<nx / 256, 256, 0, stream>>>(v, v16, nx);

  const int nw = DM_ * DM_;                          // 1,048,576
  cvt_wT<<<nw / 256, 256, 0, stream>>>(w_q,  wqT);
  cvt_wT<<<nw / 256, 256, 0, stream>>>(w_k,  wkT);
  cvt_wT<<<nw / 256, 256, 0, stream>>>(w_v,  wvT);
  cvt_wT<<<nw / 256, 256, 0, stream>>>(w_fc, wfcT);

  pack_mask<<<(B_ * S_ * 64) / 256, 256, 0, stream>>>(mask, pmask);

  proj_gemm<0><<<128, 256, 0, stream>>>(q16, wqT, qh);
  proj_gemm<0><<<128, 256, 0, stream>>>(k16, wkT, kh);
  proj_gemm<1><<<128, 256, 0, stream>>>(v16, wvT, vt);

  attn_kernel<<<512, 256, 0, stream>>>(qh, kh, vt, pmask, attn, ctx16);

  fc_gemm<<<128, 256, 0, stream>>>(ctx16, wfcT, q, xres);
  ln_kernel<<<MTOK, 256, 0, stream>>>(xres, gamma, beta, out);
}
